// MixtureOfAttention_14998025798350
// MI455X (gfx1250) — matrix operand order repaired and checked by reading, not compiled, not run on hardware
//
#include <hip/hip_runtime.h>
#include <hip/hip_bf16.h>

// ---------------------------------------------------------------------------
// Problem constants (from reference)
// ---------------------------------------------------------------------------
#define BB   4
#define NN   4096
#define DD   1024
#define GG   4
#define HH   8
#define DH   64
#define DI   512      // H*DH
#define NQ   1024
#define NK   512
#define JP   544      // padded kv length (1 null + 512 + 31 pad), multiple of 32

typedef __attribute__((ext_vector_type(16))) __bf16 v16bf;
typedef __attribute__((ext_vector_type(8)))  __bf16 v8bf;
typedef __attribute__((ext_vector_type(8)))  float  v8f;

// ---------------------------------------------------------------------------
// WMMA fragment loaders (CDNA5 wave32 layouts, CDNA5 ISA §7.12.2)
// A (16x32 bf16): lane L holds row M=L%16; halves:
//   L<16 : K = kbase+0..7  and kbase+16..23
//   L>=16: K = kbase+8..15 and kbase+24..31
// B (32x16 bf16, stored [N][K] row-major): lane L holds col N=L%16;
//   L<16 : K = kbase+0..7 and kbase+16..23 ;  L>=16: K = kbase+8..15 and kbase+24..31
// C/D (16x16 f32): lane L col N=L%16; VGPR r row M = r + 8*(L>=16)
// ---------------------------------------------------------------------------
__device__ __forceinline__ v16bf load_a_frag(const __bf16* base, int ldk,
                                             int m0, int kbase) {
  const int lane = threadIdx.x & 31;
  const int row  = lane & 15;
  const int off  = (lane >> 4) ? 8 : 0;
  const __bf16* p = base + (size_t)(m0 + row) * ldk + kbase + off;
  v8bf lo = *(const v8bf*)p;
  v8bf hi = *(const v8bf*)(p + 16);
  v16bf r;
#pragma unroll
  for (int i = 0; i < 8; ++i) { r[i] = lo[i]; r[8 + i] = hi[i]; }
  return r;
}

__device__ __forceinline__ v16bf load_b_frag(const __bf16* base, int ldk,
                                             int n0, int kbase) {
  const int lane = threadIdx.x & 31;
  const int col  = lane & 15;
  const int koff = (lane >> 4) ? 8 : 0;
  const __bf16* p = base + (size_t)(n0 + col) * ldk + kbase + koff;
  v8bf lo = *(const v8bf*)p;
  v8bf hi = *(const v8bf*)(p + 16);
  v16bf r;
#pragma unroll
  for (int i = 0; i < 8; ++i) { r[i] = lo[i]; r[8 + i] = hi[i]; }
  return r;
}

__device__ __forceinline__ v8f wmma_bf16(v16bf a, v16bf b, v8f c) {
  return __builtin_amdgcn_wmma_f32_16x16x32_bf16(false, a, false, b,
                                                 (short)0, c, false, false);
}

// Wave computes a 16(M) x 64(N) tile: C[m0:m0+16, n0:n0+64] = A*B^T
// A: [M][K] row-major bf16, B: [N][K] row-major bf16.
__device__ __forceinline__ void wave_gemm_16x64(const __bf16* A, int lda, int m0,
                                                const __bf16* B, int ldb, int n0,
                                                int K, v8f c[4]) {
  for (int k = 0; k < K; k += 32) {
    v16bf a = load_a_frag(A, lda, m0, k);
#pragma unroll
    for (int t = 0; t < 4; ++t) {
      v16bf b = load_b_frag(B, ldb, n0 + 16 * t, k);
      c[t] = wmma_bf16(a, b, c[t]);
    }
  }
}

// ---------------------------------------------------------------------------
// 1) Router: logits[b,g,n] = dot(x[b,n,:], w[g,:]) for q-router and kv-router
// ---------------------------------------------------------------------------
__global__ void router_kernel(const float* __restrict__ x,
                              const float* __restrict__ w_qr,
                              const float* __restrict__ w_kvr,
                              float* __restrict__ ql, float* __restrict__ kl) {
  const int wave = threadIdx.x >> 5;
  const int lane = threadIdx.x & 31;
  const int token = blockIdx.x * 8 + wave;           // [0, B*N)
  const int b = token >> 12;
  const int nn = token & (NN - 1);
  const float* xr = x + ((size_t)b * NN + nn) * DD;
  float acc[8] = {0.f, 0.f, 0.f, 0.f, 0.f, 0.f, 0.f, 0.f};
  for (int kk = lane; kk < DD; kk += 32) {
    float xv = xr[kk];
#pragma unroll
    for (int g = 0; g < GG; ++g) {
      acc[g]     += xv * w_qr[g * DD + kk];
      acc[4 + g] += xv * w_kvr[g * DD + kk];
    }
  }
#pragma unroll
  for (int i = 0; i < 8; ++i)
    for (int off = 16; off > 0; off >>= 1)
      acc[i] += __shfl_xor(acc[i], off, 32);
  if (lane == 0) {
#pragma unroll
    for (int g = 0; g < GG; ++g) {
      ql[((size_t)b * GG + g) * NN + nn] = acc[g];
      kl[((size_t)b * GG + g) * NN + nn] = acc[4 + g];
    }
  }
}

// ---------------------------------------------------------------------------
// 2) Top-k via full bitonic sort (descending) of 4096 (logit, idx) pairs.
//    One block per (b,g). Downstream is order-insensitive.
// ---------------------------------------------------------------------------
__global__ void topk_kernel(const float* __restrict__ logits,
                            float* __restrict__ scores,
                            int* __restrict__ idx_out, int nsel) {
  __shared__ float vals[NN];
  __shared__ int   inds[NN];
  const int bg = blockIdx.x;
  const float* L = logits + (size_t)bg * NN;
  for (int i = threadIdx.x; i < NN; i += blockDim.x) { vals[i] = L[i]; inds[i] = i; }
  __syncthreads();
  for (int k = 2; k <= NN; k <<= 1) {
    for (int j = k >> 1; j > 0; j >>= 1) {
      for (int i = threadIdx.x; i < NN; i += blockDim.x) {
        int ixj = i ^ j;
        if (ixj > i) {
          bool desc = ((i & k) == 0);
          float a = vals[i], bv = vals[ixj];
          if (desc ? (a < bv) : (a > bv)) {
            vals[i] = bv; vals[ixj] = a;
            int t = inds[i]; inds[i] = inds[ixj]; inds[ixj] = t;
          }
        }
      }
      __syncthreads();
    }
  }
  for (int i = threadIdx.x; i < nsel; i += blockDim.x) {
    float v = vals[i];
    scores[(size_t)bg * nsel + i] = 1.f / (1.f + __expf(-v));
    idx_out[(size_t)bg * nsel + i] = inds[i];
  }
}

// ---------------------------------------------------------------------------
// 3) Gather selected token rows -> bf16 matrix [bg*nsel][D]
// ---------------------------------------------------------------------------
__global__ void gather_kernel(const float* __restrict__ x,
                              const int* __restrict__ idx,
                              __bf16* __restrict__ dst, int nsel) {
  const int row = blockIdx.x;              // bg*nsel + slot
  const int bg = row / nsel;
  const int b = bg >> 2;
  const int tok = idx[row];
  const float* src = x + ((size_t)b * NN + tok) * DD;
  __bf16* d = dst + (size_t)row * DD;
  for (int c = threadIdx.x; c < DD; c += blockDim.x) d[c] = (__bf16)src[c];
}

// ---------------------------------------------------------------------------
// 4) f32 -> bf16 weight conversion
// ---------------------------------------------------------------------------
__global__ void cvt_kernel(const float* __restrict__ src,
                           __bf16* __restrict__ dst, int n) {
  int i = blockIdx.x * blockDim.x + threadIdx.x;
  if (i < n) dst[i] = (__bf16)src[i];
}

// ---------------------------------------------------------------------------
// 5) null-kv row (j=0) + zero padding (j in [513,544)) for k/v buffers.
//    kbuf layout [bgh][j][dh]; vbuf layout [bgh][dh][j] (pre-transposed for WMMA B)
// ---------------------------------------------------------------------------
__global__ void nullpad_kernel(const float* __restrict__ null_kv,
                               __bf16* __restrict__ kbuf,
                               __bf16* __restrict__ vbuf) {
  const int f = blockIdx.x;                // b*g*h
  const int g = (f >> 3) & 3;
  const int h = f & 7;
  const int dh = threadIdx.x;              // 64 threads
  __bf16* kb = kbuf + (size_t)f * JP * DH;
  __bf16* vb = vbuf + (size_t)f * DH * JP;
  kb[dh]        = (__bf16)null_kv[(((0 * GG + g) * HH + h) * 1) * DH + dh];
  vb[dh * JP]   = (__bf16)null_kv[(((1 * GG + g) * HH + h) * 1) * DH + dh];
  for (int j = NK + 1; j < JP; ++j) {
    kb[j * DH + dh] = (__bf16)0.f;
    vb[dh * JP + j] = (__bf16)0.f;
  }
}

// ---------------------------------------------------------------------------
// 6) Q projection GEMM: q[bg, m, e] = sum_d qtok[bg,m,d] * w_q[g,e,d]
//    Epilogue: store per-head bf16 layout qhead[bg][h][i][dh]
// ---------------------------------------------------------------------------
__global__ void qproj_kernel(const __bf16* __restrict__ qtok,
                             const __bf16* __restrict__ wq,
                             __bf16* __restrict__ qhead) {
  const int bg = blockIdx.z;
  const int g = bg & 3;
  const __bf16* A = qtok + (size_t)bg * NQ * DD;
  const __bf16* B = wq + (size_t)g * DI * DD;
  const int wave = threadIdx.x >> 5;
  const int m0 = blockIdx.y * 128 + wave * 16;
  const int n0 = blockIdx.x * 64;
  v8f c[4] = {};
  wave_gemm_16x64(A, DD, m0, B, DD, n0, DD, c);
  const int lane = threadIdx.x & 31;
  const int col = lane & 15;
  const int rbase = (lane >> 4) ? 8 : 0;
#pragma unroll
  for (int t = 0; t < 4; ++t) {
    int e = n0 + 16 * t + col;
    int h = e >> 6, dh = e & 63;
#pragma unroll
    for (int r = 0; r < 8; ++r) {
      int m = m0 + rbase + r;
      qhead[(((size_t)bg * HH + h) * NQ + m) * DH + dh] = (__bf16)c[t][r];
    }
  }
}

// ---------------------------------------------------------------------------
// 7) KV projection GEMM: kv[bg, j, e] = sum_d kvtok[bg,j,d] * w_kv[g,e,d]
//    e<512 -> k (layout [bgh][1+j][dh]); e>=512 -> v * kv_score (layout [bgh][dh][1+j])
// ---------------------------------------------------------------------------
__global__ void kvproj_kernel(const __bf16* __restrict__ kvtok,
                              const __bf16* __restrict__ wkv,
                              const float* __restrict__ kv_scores,
                              __bf16* __restrict__ kbuf,
                              __bf16* __restrict__ vbuf) {
  const int bg = blockIdx.z;
  const int g = bg & 3;
  const __bf16* A = kvtok + (size_t)bg * NK * DD;
  const __bf16* B = wkv + (size_t)g * (2 * DI) * DD;
  const int wave = threadIdx.x >> 5;
  const int m0 = blockIdx.y * 128 + wave * 16;
  const int n0 = blockIdx.x * 64;
  v8f c[4] = {};
  wave_gemm_16x64(A, DD, m0, B, DD, n0, DD, c);
  const int lane = threadIdx.x & 31;
  const int col = lane & 15;
  const int rbase = (lane >> 4) ? 8 : 0;
#pragma unroll
  for (int t = 0; t < 4; ++t) {
    int e = n0 + 16 * t + col;
#pragma unroll
    for (int r = 0; r < 8; ++r) {
      int m = m0 + rbase + r;                 // kv slot j in [0,512)
      float val = c[t][r];
      if (e < DI) {
        int h = e >> 6, dh = e & 63;
        kbuf[(((size_t)bg * HH + h) * JP + 1 + m) * DH + dh] = (__bf16)val;
      } else {
        int e2 = e - DI;
        int h = e2 >> 6, dh = e2 & 63;
        val *= kv_scores[(size_t)bg * NK + m];
        vbuf[(((size_t)bg * HH + h) * DH + dh) * JP + 1 + m] = (__bf16)val;
      }
    }
  }
}

// ---------------------------------------------------------------------------
// 8) Flash attention: one wave per 16-row Q tile, online softmax over 513 KV
//    (padded to JP=544), WMMA for both S=Q*K^T and O=P*V.
// ---------------------------------------------------------------------------
__global__ void attn_kernel(const __bf16* __restrict__ qhead,
                            const __bf16* __restrict__ kbuf,
                            const __bf16* __restrict__ vbuf,
                            __bf16* __restrict__ obuf) {
  __shared__ __bf16 pbuf[4 * 16 * 48];       // per-wave P scratch (ld=48, 16B rows)
  const int flat = blockIdx.x;               // b*g*h
  const int h = flat & 7;
  const int bg = flat >> 3;
  const int wave = threadIdx.x >> 5;
  const int lane = threadIdx.x & 31;
  const int col = lane & 15;
  const int rbase = (lane >> 4) ? 8 : 0;
  const int i0 = (blockIdx.y * 4 + wave) * 16;

  const __bf16* qb = qhead + (size_t)flat * NQ * DH;   // [i][dh]
  const __bf16* kb = kbuf + (size_t)flat * JP * DH;    // [j][dh]
  const __bf16* vb = vbuf + (size_t)flat * DH * JP;    // [dh][j]
  __bf16* pw = pbuf + wave * 16 * 48;

  v16bf qa0 = load_a_frag(qb, DH, i0, 0);
  v16bf qa1 = load_a_frag(qb, DH, i0, 32);

  float mrow[8], lrow[8];
  v8f o[4] = {};
#pragma unroll
  for (int r = 0; r < 8; ++r) { mrow[r] = -3.0e38f; lrow[r] = 0.f; }

  const float scale = 0.125f;                // DH^-0.5
  for (int jt = 0; jt < JP / 32; ++jt) {
    const int j0 = jt * 32;
    v8f s0 = {}, s1 = {};
    s0 = wmma_bf16(qa0, load_b_frag(kb, DH, j0, 0), s0);
    s0 = wmma_bf16(qa1, load_b_frag(kb, DH, j0, 32), s0);
    s1 = wmma_bf16(qa0, load_b_frag(kb, DH, j0 + 16, 0), s1);
    s1 = wmma_bf16(qa1, load_b_frag(kb, DH, j0 + 16, 32), s1);

    const bool ok0 = (j0 + col) < (NK + 1);
    const bool ok1 = (j0 + 16 + col) < (NK + 1);
#pragma unroll
    for (int r = 0; r < 8; ++r) {
      float p0 = ok0 ? s0[r] * scale : -1.0e30f;
      float p1 = ok1 ? s1[r] * scale : -1.0e30f;
      float mx = fmaxf(p0, p1);
#pragma unroll
      for (int off = 1; off < 16; off <<= 1) mx = fmaxf(mx, __shfl_xor(mx, off, 32));
      float mnew = fmaxf(mrow[r], mx);
      float alpha = __expf(mrow[r] - mnew);
      float e0 = __expf(p0 - mnew);
      float e1 = __expf(p1 - mnew);
      float sm = e0 + e1;
#pragma unroll
      for (int off = 1; off < 16; off <<= 1) sm += __shfl_xor(sm, off, 32);
      lrow[r] = lrow[r] * alpha + sm;
      mrow[r] = mnew;
      o[0][r] *= alpha; o[1][r] *= alpha; o[2][r] *= alpha; o[3][r] *= alpha;
      pw[(rbase + r) * 48 + col]      = (__bf16)e0;
      pw[(rbase + r) * 48 + 16 + col] = (__bf16)e1;
    }
    asm volatile("s_wait_dscnt 0" ::: "memory");   // P stores visible to A-frag loads
    v16bf pa = load_a_frag(pw, 48, 0, 0);
#pragma unroll
    for (int t = 0; t < 4; ++t) {
      v16bf vfr = load_b_frag(vb, JP, 16 * t, j0);
      o[t] = wmma_bf16(pa, vfr, o[t]);
    }
  }
#pragma unroll
  for (int t = 0; t < 4; ++t)
#pragma unroll
    for (int r = 0; r < 8; ++r) {
      float val = o[t][r] / lrow[r];
      int i = i0 + rbase + r;
      obuf[((size_t)bg * NQ + i) * DI + h * DH + 16 * t + col] = (__bf16)val;
    }
}

// ---------------------------------------------------------------------------
// 9) Out projection GEMM + fused q_score scale + scatter-add (atomic)
//    o[bg, m, dcol] = sum_e obuf[bg,m,e] * w_out[g,dcol,e]
// ---------------------------------------------------------------------------
__global__ void outproj_kernel(const __bf16* __restrict__ obuf,
                               const __bf16* __restrict__ wout,
                               const float* __restrict__ q_scores,
                               const int* __restrict__ q_idx,
                               float* __restrict__ summed) {
  const int bg = blockIdx.z;
  const int g = bg & 3;
  const int b = bg >> 2;
  const __bf16* A = obuf + (size_t)bg * NQ * DI;
  const __bf16* B = wout + (size_t)g * DD * DI;
  const int wave = threadIdx.x >> 5;
  const int m0 = blockIdx.y * 128 + wave * 16;
  const int n0 = blockIdx.x * 64;
  v8f c[4] = {};
  wave_gemm_16x64(A, DI, m0, B, DI, n0, DI, c);
  const int lane = threadIdx.x & 31;
  const int col = lane & 15;
  const int rbase = (lane >> 4) ? 8 : 0;
#pragma unroll
  for (int r = 0; r < 8; ++r) {
    int m = m0 + rbase + r;
    float qs = q_scores[(size_t)bg * NQ + m];
    int tok = q_idx[(size_t)bg * NQ + m];
    float* dstrow = summed + ((size_t)b * NN + tok) * DD;
#pragma unroll
    for (int t = 0; t < 4; ++t) {
      int n = n0 + 16 * t + col;
      atomicAdd(&dstrow[n], c[t][r] * qs);
    }
  }
}

// ---------------------------------------------------------------------------
// 10) counts[b, tok] += 1 for every selected query slot
// ---------------------------------------------------------------------------
__global__ void counts_kernel(const int* __restrict__ q_idx,
                              float* __restrict__ counts) {
  int i = blockIdx.x * blockDim.x + threadIdx.x;
  if (i < BB * GG * NQ) {
    int b = i / (GG * NQ);
    atomicAdd(&counts[(size_t)b * NN + q_idx[i]], 1.0f);
  }
}

// ---------------------------------------------------------------------------
// 11) finalize: out = counts>0 ? summed/max(counts,1e-5) : null_token
// ---------------------------------------------------------------------------
__global__ void finalize_kernel(const float* __restrict__ summed,
                                const float* __restrict__ counts,
                                const float* __restrict__ null_token,
                                float* __restrict__ out) {
  size_t i = (size_t)blockIdx.x * blockDim.x + threadIdx.x;
  if (i >= (size_t)BB * NN * DD) return;
  int dcol = (int)(i & (DD - 1));
  size_t bn = i / DD;
  float cnt = counts[bn];
  out[i] = (cnt > 0.f) ? summed[i] / fmaxf(cnt, 1e-5f) : null_token[dcol];
}

// ---------------------------------------------------------------------------
// Launch
// ---------------------------------------------------------------------------
static inline char* carve(char*& p, size_t bytes) {
  char* r = p;
  p += (bytes + 255) & ~(size_t)255;
  return r;
}

extern "C" void kernel_launch(void* const* d_in, const int* in_sizes, int n_in,
                              void* d_out, int out_size, void* d_ws, size_t ws_size,
                              hipStream_t stream) {
  const float* x      = (const float*)d_in[0];
  const float* w_qr   = (const float*)d_in[1];
  const float* w_kvr  = (const float*)d_in[2];
  const float* w_q    = (const float*)d_in[3];
  const float* w_kv   = (const float*)d_in[4];
  const float* w_out  = (const float*)d_in[5];
  const float* nullkv = (const float*)d_in[6];
  const float* nulltk = (const float*)d_in[7];
  float* out = (float*)d_out;

  char* p = (char*)d_ws;
  float* ql       = (float*)carve(p, (size_t)BB * GG * NN * 4);
  float* kl       = (float*)carve(p, (size_t)BB * GG * NN * 4);
  float* q_scores = (float*)carve(p, (size_t)BB * GG * NQ * 4);
  int*   q_idx    = (int*)  carve(p, (size_t)BB * GG * NQ * 4);
  float* kv_scores= (float*)carve(p, (size_t)BB * GG * NK * 4);
  int*   kv_idx   = (int*)  carve(p, (size_t)BB * GG * NK * 4);
  __bf16* wq_bf   = (__bf16*)carve(p, (size_t)GG * DI * DD * 2);
  __bf16* wkv_bf  = (__bf16*)carve(p, (size_t)GG * 2 * DI * DD * 2);
  __bf16* wout_bf = (__bf16*)carve(p, (size_t)GG * DD * DI * 2);
  __bf16* qtok    = (__bf16*)carve(p, (size_t)BB * GG * NQ * DD * 2);
  __bf16* kvtok   = (__bf16*)carve(p, (size_t)BB * GG * NK * DD * 2);
  __bf16* qhead   = (__bf16*)carve(p, (size_t)BB * GG * HH * NQ * DH * 2);
  __bf16* kbuf    = (__bf16*)carve(p, (size_t)BB * GG * HH * JP * DH * 2);
  __bf16* vbuf    = (__bf16*)carve(p, (size_t)BB * GG * HH * DH * JP * 2);
  __bf16* obuf    = (__bf16*)carve(p, (size_t)BB * GG * NQ * DI * 2);
  float* summed   = (float*)carve(p, (size_t)BB * NN * DD * 4);
  float* counts   = (float*)carve(p, (size_t)BB * NN * 4);

  hipMemsetAsync(summed, 0, (size_t)BB * NN * DD * 4, stream);
  hipMemsetAsync(counts, 0, (size_t)BB * NN * 4, stream);

  // weight conversions
  cvt_kernel<<<(GG * DI * DD + 255) / 256, 256, 0, stream>>>(w_q, wq_bf, GG * DI * DD);
  cvt_kernel<<<(GG * 2 * DI * DD + 255) / 256, 256, 0, stream>>>(w_kv, wkv_bf, GG * 2 * DI * DD);
  cvt_kernel<<<(GG * DD * DI + 255) / 256, 256, 0, stream>>>(w_out, wout_bf, GG * DD * DI);

  // routing
  router_kernel<<<BB * NN / 8, 256, 0, stream>>>(x, w_qr, w_kvr, ql, kl);
  topk_kernel<<<BB * GG, 512, 0, stream>>>(ql, q_scores, q_idx, NQ);
  topk_kernel<<<BB * GG, 512, 0, stream>>>(kl, kv_scores, kv_idx, NK);

  // gather selected tokens (bf16)
  gather_kernel<<<BB * GG * NQ, 256, 0, stream>>>(x, q_idx, qtok, NQ);
  gather_kernel<<<BB * GG * NK, 256, 0, stream>>>(x, kv_idx, kvtok, NK);

  // null kv + pad
  nullpad_kernel<<<BB * GG * HH, 64, 0, stream>>>(nullkv, kbuf, vbuf);

  // projections (WMMA GEMMs)
  qproj_kernel<<<dim3(DI / 64, NQ / 128, BB * GG), 256, 0, stream>>>(qtok, wq_bf, qhead);
  kvproj_kernel<<<dim3(2 * DI / 64, NK / 128, BB * GG), 256, 0, stream>>>(kvtok, wkv_bf, kv_scores, kbuf, vbuf);

  // attention (WMMA flash)
  attn_kernel<<<dim3(BB * GG * HH, NQ / 64), 128, 0, stream>>>(qhead, kbuf, vbuf, obuf);

  // output projection + scatter-add
  outproj_kernel<<<dim3(DD / 64, NQ / 128, BB * GG), 256, 0, stream>>>(obuf, wout_bf, q_scores, q_idx, summed);
  counts_kernel<<<(BB * GG * NQ + 255) / 256, 256, 0, stream>>>(q_idx, counts);

  // finalize
  finalize_kernel<<<(int)(((size_t)BB * NN * DD + 255) / 256), 256, 0, stream>>>(summed, counts, nulltk, out);
}
